// ScaledDotProductAttention_28802050687501
// MI455X (gfx1250) — matrix operand order repaired and checked by reading, not compiled, not run on hardware
//
#include <hip/hip_runtime.h>
#include <hip/hip_bf16.h>

// ===========================================================================
// Fused attention for MI455X (gfx1250), all GEMMs on v_wmma_f32_16x16x32_bf16.
//
// Pipeline (B=8, L=1024, D=1024):
//   proj_kernel<0..2> : Q = (X Wq)/32 [bf16], K = X Wk [bf16], Vt = (X Wv)^T
//   qk_kernel         : S = Q K^T causal (async double-buffered LDS) [f32, ws]
//   softmax_kernel    : P = softmax(S) rows (zeros above diagonal) [bf16, ws]
//   pv_kernel         : O = P V (k-loop clipped at diagonal, async LDS) -> d_out
//
// S (32MB) + P (16MB) stay resident in the 192MB L2.  Workspace: 96 MB.
// ===========================================================================

typedef __bf16 bf16;
typedef bf16  v8bf  __attribute__((ext_vector_type(8)));
typedef bf16  v16bf __attribute__((ext_vector_type(16)));
typedef float v8f   __attribute__((ext_vector_type(8)));

#define BM 128          // block tile M
#define BN 128          // block tile N
#define BK 32           // k step (matches wmma 16x16x32)
#define KP 40           // padded LDS row stride in bf16 (80B, 16B-aligned, conflict-free)

// ---- gfx1250 async global->LDS (ASYNCcnt) with safe fallback ---------------
#if __has_builtin(__builtin_amdgcn_global_load_async_to_lds_b128) && \
    __has_builtin(__builtin_amdgcn_s_wait_asynccnt)
#define HAVE_ASYNC_LDS 1
#else
#define HAVE_ASYNC_LDS 0
#endif

// Builtin param types (from clang diagnostic): GCC-vector int4 pointers,
// addrspace(1) source / addrspace(3) destination.
typedef int v4i_gcc __attribute__((vector_size(4 * sizeof(int))));
typedef __attribute__((address_space(1))) v4i_gcc as1_v4i;
typedef __attribute__((address_space(3))) v4i_gcc as3_v4i;

static __device__ __forceinline__ void async_copy16(const bf16* g, bf16* l) {
#if HAVE_ASYNC_LDS
    __builtin_amdgcn_global_load_async_to_lds_b128((as1_v4i*)g, (as3_v4i*)l, 0, 0);
#else
    *(v8bf*)l = *(const v8bf*)g;
#endif
}
static __device__ __forceinline__ void async_wait4() {
#if HAVE_ASYNC_LDS
    __builtin_amdgcn_s_wait_asynccnt(4);   // 4 newest (next tile) may remain
#endif
}
static __device__ __forceinline__ void async_wait0() {
#if HAVE_ASYNC_LDS
    __builtin_amdgcn_s_wait_asynccnt(0);
#endif
}

// ---- fragment helpers (layouts per CDNA5 ISA 7.12.2, wave32) ---------------

static __device__ __forceinline__ v16bf combine16(v8bf lo, v8bf hi) {
    v16bf r;
#pragma unroll
    for (int i = 0; i < 8; ++i) { r[i] = lo[i]; r[i + 8] = hi[i]; }
    return r;
}

// A-matrix 16x32 bf16: lane l -> row (l&15); elems 0..7 = K base0..base0+7,
// elems 8..15 = K 16+base0..+7, base0 = (l>>4)*8.
static __device__ __forceinline__ v16bf load_a_frag(const bf16* As, int m_base, int lane) {
    const int  row = m_base + (lane & 15);
    const int  b0  = (lane >> 4) << 3;
    const bf16* p  = As + row * KP;
    v8bf lo = *(const v8bf*)(p + b0);
    v8bf hi = *(const v8bf*)(p + 16 + b0);
    return combine16(lo, hi);
}

// B-matrix 32x16 bf16 in LDS as [n][k]: lane l -> col (l&15);
// elems 0..7 = K base0..base0+7, elems 8..15 = K 16+base0..+7, base0 = (l>>4)*8.
static __device__ __forceinline__ v16bf load_b_frag(const bf16* Bs, int n_base, int lane) {
    const int  col = n_base + (lane & 15);
    const bf16* p  = Bs + col * KP + ((lane >> 4) << 3);
    v8bf lo = *(const v8bf*)(p);
    v8bf hi = *(const v8bf*)(p + 16);
    return combine16(lo, hi);
}

static __device__ __forceinline__ v8bf cvt8(float4 a, float4 b) {
    v8bf r;
    r[0] = (bf16)a.x; r[1] = (bf16)a.y; r[2] = (bf16)a.z; r[3] = (bf16)a.w;
    r[4] = (bf16)b.x; r[5] = (bf16)b.y; r[6] = (bf16)b.z; r[7] = (bf16)b.w;
    return r;
}

#define WMMA_BF16(ACC, A, B) \
    (ACC) = __builtin_amdgcn_wmma_f32_16x16x32_bf16(false, (A), false, (B), (short)0, (ACC), false, false)

// ===========================================================================
// Kernel 1: projection GEMM  Out = X @ W  (f32 in -> bf16 out)
//   MODE 0: Q (scaled by 1/32)   MODE 1: K   MODE 2: V^T per batch [b][d][l]
// ===========================================================================
template <int MODE>
__global__ __launch_bounds__(256)
void proj_kernel(const float* __restrict__ X, const float* __restrict__ W,
                 bf16* __restrict__ Out) {
    __shared__ bf16 As[BM * KP];
    __shared__ bf16 Bs[BN * KP];

    const int tid  = threadIdx.x;
    const int lane = tid & 31;
    const int wave = tid >> 5;
    const int wm   = (wave >> 2) * 64;
    const int wn   = (wave & 3) * 32;
    const int m0   = blockIdx.y * BM;
    const int n0   = blockIdx.x * BN;

    v8f acc[4][2];
#pragma unroll
    for (int i = 0; i < 4; ++i)
#pragma unroll
        for (int j = 0; j < 2; ++j)
#pragma unroll
            for (int e = 0; e < 8; ++e) acc[i][j][e] = 0.0f;

    const int arow = tid >> 1;           // 0..127
    const int acol = (tid & 1) << 4;     // 0 or 16
    const int bk   = tid & 31;           // W tile row (k)
    const int bnl  = (tid >> 5) << 4;    // W tile col group

    for (int k0 = 0; k0 < 1024; k0 += BK) {
        const float* ap = X + (size_t)(m0 + arow) * 1024 + k0 + acol;
        float4 a0 = *(const float4*)(ap);
        float4 a1 = *(const float4*)(ap + 4);
        float4 a2 = *(const float4*)(ap + 8);
        float4 a3 = *(const float4*)(ap + 12);
        bf16* as = As + arow * KP + acol;
        *(v8bf*)(as)     = cvt8(a0, a1);
        *(v8bf*)(as + 8) = cvt8(a2, a3);

        const float* wp = W + (size_t)(k0 + bk) * 1024 + n0 + bnl;
        float4 w0 = *(const float4*)(wp);
        float4 w1 = *(const float4*)(wp + 4);
        float4 w2 = *(const float4*)(wp + 8);
        float4 w3 = *(const float4*)(wp + 12);
        float wv[16] = { w0.x, w0.y, w0.z, w0.w, w1.x, w1.y, w1.z, w1.w,
                         w2.x, w2.y, w2.z, w2.w, w3.x, w3.y, w3.z, w3.w };
        bf16* bs = Bs + bnl * KP + bk;
#pragma unroll
        for (int i = 0; i < 16; ++i) bs[i * KP] = (bf16)wv[i];

        __syncthreads();
        v16bf b0f = load_b_frag(Bs, wn, lane);
        v16bf b1f = load_b_frag(Bs, wn + 16, lane);
#pragma unroll
        for (int i = 0; i < 4; ++i) {
            v16bf af = load_a_frag(As, wm + 16 * i, lane);
            WMMA_BF16(acc[i][0], af, b0f);
            WMMA_BF16(acc[i][1], af, b1f);
        }
        __syncthreads();
    }

    const int cn = lane & 15;
    const int rs = (lane >> 4) << 3;
    if (MODE == 2) {
        // V^T: [b][d][l]; tiles never straddle batches (1024 % 128 == 0).
        // Per lane the 8 row-elements are contiguous in l -> one b128 store.
        bf16* vbase = Out + (size_t)(m0 >> 10) * 1048576 + (m0 & 1023) + wm + rs;
#pragma unroll
        for (int j = 0; j < 2; ++j)
#pragma unroll
            for (int i = 0; i < 4; ++i) {
                bf16* p = vbase + (size_t)(n0 + wn + 16 * j + cn) * 1024 + 16 * i;
                v8bf o;
#pragma unroll
                for (int e = 0; e < 8; ++e) o[e] = (bf16)acc[i][j][e];
                *(v8bf*)p = o;
            }
    } else {
#pragma unroll
        for (int i = 0; i < 4; ++i) {
            bf16* p = Out + (size_t)(m0 + wm + 16 * i + rs) * 1024 + n0 + wn + cn;
#pragma unroll
            for (int j = 0; j < 2; ++j)
#pragma unroll
                for (int e = 0; e < 8; ++e) {
                    float v = acc[i][j][e];
                    p[(size_t)e * 1024 + 16 * j] =
                        (bf16)(MODE == 0 ? v * 0.03125f : v);   // fold 1/sqrt(D)
                }
        }
    }
}

// ===========================================================================
// Kernel 2: S = Q K^T (Q pre-scaled), causal; async double-buffered LDS.
// ===========================================================================
__global__ __launch_bounds__(256)
void qk_kernel(const bf16* __restrict__ Q, const bf16* __restrict__ K,
               float* __restrict__ S) {
    const int b  = blockIdx.z;
    const int m0 = blockIdx.y * BM;
    const int n0 = blockIdx.x * BN;
    if (n0 > m0 + (BM - 1)) return;          // uniform: tile fully above diagonal

    __shared__ bf16 As[2][BM * KP];
    __shared__ bf16 Bs[2][BN * KP];

    const int tid  = threadIdx.x;
    const int lane = tid & 31;
    const int wave = tid >> 5;
    const int wm   = (wave >> 2) * 64;
    const int wn   = (wave & 3) * 32;

    v8f acc[4][2];
#pragma unroll
    for (int i = 0; i < 4; ++i)
#pragma unroll
        for (int j = 0; j < 2; ++j)
#pragma unroll
            for (int e = 0; e < 8; ++e) acc[i][j][e] = 0.0f;

    const int arow = tid >> 1;
    const int acol = (tid & 1) << 4;
    const bf16* qrow = Q + (size_t)b * 1048576 + (size_t)(m0 + arow) * 1024 + acol;
    const bf16* krow = K + (size_t)b * 1048576 + (size_t)(n0 + arow) * 1024 + acol;

    auto issue = [&](int kt, int buf) {
        const bf16* qp = qrow + kt * BK;
        const bf16* kp = krow + kt * BK;
        bf16* as = &As[buf][arow * KP + acol];
        bf16* bs = &Bs[buf][arow * KP + acol];
        async_copy16(qp,     as);
        async_copy16(qp + 8, as + 8);
        async_copy16(kp,     bs);
        async_copy16(kp + 8, bs + 8);
    };
    auto compute = [&](int buf) {
        v16bf b0f = load_b_frag(&Bs[buf][0], wn, lane);
        v16bf b1f = load_b_frag(&Bs[buf][0], wn + 16, lane);
#pragma unroll
        for (int i = 0; i < 4; ++i) {
            v16bf af = load_a_frag(&As[buf][0], wm + 16 * i, lane);
            WMMA_BF16(acc[i][0], af, b0f);
            WMMA_BF16(acc[i][1], af, b1f);
        }
    };

    const int nk = 1024 / BK;
    issue(0, 0);
    for (int kt = 0; kt < nk - 1; ++kt) {
        issue(kt + 1, (kt + 1) & 1);
        async_wait4();
        __syncthreads();
        compute(kt & 1);
        __syncthreads();
    }
    async_wait0();
    __syncthreads();
    compute((nk - 1) & 1);

    float* Sb = S + (size_t)b * 1048576;
    const int cn = lane & 15;
    const int rs = (lane >> 4) << 3;
#pragma unroll
    for (int i = 0; i < 4; ++i) {
        const int r0 = m0 + wm + 16 * i + rs;
        float* p = Sb + (size_t)r0 * 1024 + n0 + wn + cn;
#pragma unroll
        for (int j = 0; j < 2; ++j) {
            const int c = n0 + wn + 16 * j + cn;
#pragma unroll
            for (int e = 0; e < 8; ++e)
                p[(size_t)e * 1024 + 16 * j] =
                    (c <= r0 + e) ? acc[i][j][e] : -__builtin_inff();
        }
    }
}

// ===========================================================================
// Kernel 3: row softmax, S[f32] -> P[bf16]; zeros above diagonal.
// ===========================================================================
static __device__ __forceinline__ float wave_max(float v) {
#pragma unroll
    for (int off = 16; off > 0; off >>= 1) v = fmaxf(v, __shfl_xor(v, off, 32));
    return v;
}
static __device__ __forceinline__ float wave_sum(float v) {
#pragma unroll
    for (int off = 16; off > 0; off >>= 1) v += __shfl_xor(v, off, 32);
    return v;
}

__global__ __launch_bounds__(256)
void softmax_kernel(const float* __restrict__ S, bf16* __restrict__ P) {
    const int row = blockIdx.x;          // b*1024 + q
    const int q   = row & 1023;
    const int len = q + 1;
    const float* s = S + (size_t)row * 1024;
    bf16*        p = P + (size_t)row * 1024;

    const int tid  = threadIdx.x;
    const int lane = tid & 31;
    const int wave = tid >> 5;

    __shared__ float redm[8];
    __shared__ float reds[8];

    float lm = -__builtin_inff();
    for (int j = tid; j < len; j += 256) lm = fmaxf(lm, s[j]);
    lm = wave_max(lm);
    if (lane == 0) redm[wave] = lm;
    __syncthreads();
    float m = -__builtin_inff();
#pragma unroll
    for (int i = 0; i < 8; ++i) m = fmaxf(m, redm[i]);

    float ls = 0.0f;
    for (int j = tid; j < len; j += 256) ls += __expf(s[j] - m);
    ls = wave_sum(ls);
    if (lane == 0) reds[wave] = ls;
    __syncthreads();
    float sum = 0.0f;
#pragma unroll
    for (int i = 0; i < 8; ++i) sum += reds[i];
    const float inv = 1.0f / sum;

    for (int j = tid; j < 1024; j += 256)
        p[j] = (j < len) ? (bf16)(__expf(s[j] - m) * inv) : (bf16)0.0f;
}

// ===========================================================================
// Kernel 4: O = P @ V  (A = P bf16, B = V^T[d][key] bf16), async LDS pipeline,
// k-loop clipped at diagonal (P is zero above it).
// ===========================================================================
__global__ __launch_bounds__(256)
void pv_kernel(const bf16* __restrict__ P, const bf16* __restrict__ Vt,
               float* __restrict__ O) {
    const int b  = blockIdx.z;
    const int m0 = blockIdx.y * BM;      // query tile
    const int n0 = blockIdx.x * BN;      // d tile

    __shared__ bf16 As[2][BM * KP];
    __shared__ bf16 Bs[2][BN * KP];

    const int tid  = threadIdx.x;
    const int lane = tid & 31;
    const int wave = tid >> 5;
    const int wm   = (wave >> 2) * 64;
    const int wn   = (wave & 3) * 32;

    v8f acc[4][2];
#pragma unroll
    for (int i = 0; i < 4; ++i)
#pragma unroll
        for (int j = 0; j < 2; ++j)
#pragma unroll
            for (int e = 0; e < 8; ++e) acc[i][j][e] = 0.0f;

    const int arow = tid >> 1;
    const int acol = (tid & 1) << 4;
    const bf16* prow = P  + (size_t)b * 1048576 + (size_t)(m0 + arow) * 1024 + acol;
    const bf16* vrow = Vt + (size_t)b * 1048576 + (size_t)(n0 + arow) * 1024 + acol;

    auto issue = [&](int kt, int buf) {
        const bf16* pp = prow + kt * BK;
        const bf16* vp = vrow + kt * BK;
        bf16* as = &As[buf][arow * KP + acol];
        bf16* bs = &Bs[buf][arow * KP + acol];
        async_copy16(pp,     as);
        async_copy16(pp + 8, as + 8);
        async_copy16(vp,     bs);
        async_copy16(vp + 8, bs + 8);
    };
    auto compute = [&](int buf) {
        v16bf b0f = load_b_frag(&Bs[buf][0], wn, lane);
        v16bf b1f = load_b_frag(&Bs[buf][0], wn + 16, lane);
#pragma unroll
        for (int i = 0; i < 4; ++i) {
            v16bf af = load_a_frag(&As[buf][0], wm + 16 * i, lane);
            WMMA_BF16(acc[i][0], af, b0f);
            WMMA_BF16(acc[i][1], af, b1f);
        }
    };

    const int nk = (m0 + BM) / BK;       // causal clip: keys <= m0+127
    issue(0, 0);
    for (int kt = 0; kt < nk - 1; ++kt) {
        issue(kt + 1, (kt + 1) & 1);
        async_wait4();
        __syncthreads();
        compute(kt & 1);
        __syncthreads();
    }
    async_wait0();
    __syncthreads();
    compute((nk - 1) & 1);

    float* Ob = O + (size_t)b * 1048576;
    const int cn = lane & 15;
    const int rs = (lane >> 4) << 3;
#pragma unroll
    for (int i = 0; i < 4; ++i) {
        float* p = Ob + (size_t)(m0 + wm + 16 * i + rs) * 1024 + n0 + wn + cn;
#pragma unroll
        for (int j = 0; j < 2; ++j)
#pragma unroll
            for (int e = 0; e < 8; ++e)
                p[(size_t)e * 1024 + 16 * j] = acc[i][j][e];
    }
}

// ===========================================================================
extern "C" void kernel_launch(void* const* d_in, const int* in_sizes, int n_in,
                              void* d_out, int out_size, void* d_ws, size_t ws_size,
                              hipStream_t stream) {
    (void)in_sizes; (void)n_in; (void)out_size; (void)ws_size;

    const float* query = (const float*)d_in[0];
    const float* key   = (const float*)d_in[1];
    const float* value = (const float*)d_in[2];
    const float* Wq    = (const float*)d_in[3];
    const float* Wk    = (const float*)d_in[4];
    const float* Wv    = (const float*)d_in[5];
    float* out = (float*)d_out;

    // Workspace layout (96 MB): Q|K|Vt bf16 (16MB each), S f32 (32MB), P bf16 (16MB)
    const size_t NELEM = (size_t)8 * 1024 * 1024;
    bf16*  Qb = (bf16*)d_ws;
    bf16*  Kb = Qb + NELEM;
    bf16*  Vt = Kb + NELEM;
    float* S  = (float*)(Vt + NELEM);
    bf16*  Pp = (bf16*)(S + NELEM);

    const dim3 blk(256);
    const dim3 gproj(1024 / BN, 8192 / BM, 1);   // (8, 64)
    proj_kernel<0><<<gproj, blk, 0, stream>>>(query, Wq, Qb);
    proj_kernel<1><<<gproj, blk, 0, stream>>>(key,   Wk, Kb);
    proj_kernel<2><<<gproj, blk, 0, stream>>>(value, Wv, Vt);

    const dim3 gs(8, 8, 8);                      // (n_tiles, m_tiles, batch)
    qk_kernel<<<gs, blk, 0, stream>>>(Qb, Kb, S);

    softmax_kernel<<<dim3(8192), blk, 0, stream>>>(S, Pp);

    pv_kernel<<<gs, blk, 0, stream>>>(Pp, Vt, out);
}
